// AlignmentLoss_7885559955748
// MI455X (gfx1250) — matrix operand order repaired and checked by reading, not compiled, not run on hardware
//
#include <hip/hip_runtime.h>
#include <hip/hip_bf16.h>

// ---------------------------------------------------------------------------
// AlignmentLoss on MI455X (gfx1250, wave32, WMMA):
//   G = X @ X^T  (X: 4096x1024 f32), reduce strictly-lower triangle:
//     s1 = sum_{i>j} G[i,j]*(t_i==t_j),  s2 = sum_{i>j} (t_i==t_j),
//     s3 = sum_{i>j} G[i,j]^2
//   out = -( s1 / (s2 * sqrt(s3)) )
// Split-bf16 GEMM (hi*hi + hi*lo + lo*hi) on v_wmma_f32_16x16x32_bf16 over the
// 528 lower-triangular 128x128 tiles; masks fused into the epilogue.
// Data movement: double-buffered LDS stages filled with
// GLOBAL_LOAD_ASYNC_TO_LDS_B128 (ASYNCcnt), overlapping prefetch with WMMA.
// ---------------------------------------------------------------------------

#define USE_ASYNC_LDS 1

typedef __attribute__((ext_vector_type(16))) __bf16 v16bf;
typedef __attribute__((ext_vector_type(8)))  float  v8f;

static constexpr int Nrows = 4096;
static constexpr int Dcols = 1024;
static constexpr int NB    = Nrows / 128;           // 32 block-rows
static constexpr int NTILE = NB * (NB + 1) / 2;     // 528 lower-tri tiles
static constexpr int KC    = 32;                    // K chunk per LDS stage
static constexpr int NK    = Dcols / KC;            // 32 chunks
static constexpr int LDR   = 40;                    // padded LDS row (bf16) = 80B
static constexpr int PANEL = 128 * LDR;             // bf16 elems per panel

// ---------------------------------------------------------------------------
// Kernel 1: f32 -> bf16 hi + bf16 lo residual split (4 elems / thread)
// ---------------------------------------------------------------------------
__global__ __launch_bounds__(256)
void convert_split(const float* __restrict__ x,
                   __bf16* __restrict__ xhi, __bf16* __restrict__ xlo) {
    size_t g = (size_t)blockIdx.x * blockDim.x + threadIdx.x;   // float4 index
    const float4 v = ((const float4*)x)[g];
    union { __bf16 h[4]; ushort4 u; } hi, lo;
    float f[4] = {v.x, v.y, v.z, v.w};
#pragma unroll
    for (int k = 0; k < 4; ++k) {
        __bf16 h = (__bf16)f[k];
        hi.h[k] = h;
        lo.h[k] = (__bf16)(f[k] - (float)h);
    }
    ((ushort4*)xhi)[g] = hi.u;
    ((ushort4*)xlo)[g] = lo.u;
}

// ---------------------------------------------------------------------------
// Async stage fill: per lane, 4 panels x 2 x 16B direct global->LDS copies.
// SADDR form: global addr = SGPR64 base + VGPR32 byte offset (+ inst offset,
// which is added to BOTH the global and LDS addresses per the ISA).
// ---------------------------------------------------------------------------
#if USE_ASYNC_LDS
__device__ __forceinline__ void async_stage(const __bf16* xhi, const __bf16* xlo,
                                            unsigned gA, unsigned gB,
                                            unsigned lAhi, unsigned lAlo,
                                            unsigned lBhi, unsigned lBlo) {
    asm volatile(
        "global_load_async_to_lds_b128 %0, %4, %6\n\t"
        "global_load_async_to_lds_b128 %0, %4, %6 offset:16\n\t"
        "global_load_async_to_lds_b128 %1, %4, %7\n\t"
        "global_load_async_to_lds_b128 %1, %4, %7 offset:16\n\t"
        "global_load_async_to_lds_b128 %2, %5, %6\n\t"
        "global_load_async_to_lds_b128 %2, %5, %6 offset:16\n\t"
        "global_load_async_to_lds_b128 %3, %5, %7\n\t"
        "global_load_async_to_lds_b128 %3, %5, %7 offset:16"
        :
        : "v"(lAhi), "v"(lAlo), "v"(lBhi), "v"(lBlo),
          "v"(gA), "v"(gB), "s"(xhi), "s"(xlo)
        : "memory");
}
#endif

// ---------------------------------------------------------------------------
// Kernel 2: one 128x128 lower-triangular tile of G per workgroup (8 waves).
// Wave w owns a 32x64 sub-tile = 2x4 WMMA accumulators.
// ---------------------------------------------------------------------------
__global__ __launch_bounds__(256)
void gram_tril_wmma(const __bf16* __restrict__ xhi,
                    const __bf16* __restrict__ xlo,
                    const int*    __restrict__ tgt,
                    float*        __restrict__ partials) {
    // 2 stages x 4 panels (Ahi, Alo, Bhi, Blo) x 128x32 bf16 (padded) = 80 KB
    __shared__ __align__(16) __bf16 smem[2][4][PANEL];
    __shared__ float sAcc[3];

    // --- decode linear tile id -> (bi, bj) with bi >= bj ---
    const unsigned t = blockIdx.x;
    unsigned bi = (unsigned)((sqrtf(8.0f * (float)t + 1.0f) - 1.0f) * 0.5f);
    while ((bi + 1u) * (bi + 2u) / 2u <= t) ++bi;
    while (bi * (bi + 1u) / 2u > t) --bi;
    const unsigned bj = t - bi * (bi + 1u) / 2u;

    const int tid  = threadIdx.x;
    const int wave = tid >> 5;
    const int lane = tid & 31;
    const int n16  = lane & 15;       // N index within a 16x16 tile
    const int lh   = lane >> 4;       // lane-half (0: lanes 0-15, 1: lanes 16-31)
    const int wRow = (wave >> 1) * 32;   // 4 wave-rows of 32
    const int wCol = (wave & 1) * 64;    // 2 wave-cols of 64

    if (tid < 3) sAcc[tid] = 0.0f;

    v8f acc[2][4];
#pragma unroll
    for (int mt = 0; mt < 2; ++mt)
#pragma unroll
        for (int nt = 0; nt < 4; ++nt) acc[mt][nt] = (v8f)0.0f;

    // cooperative-load coordinates: 256 threads x 32B per panel per stage
    const int ldRow = tid >> 1;            // 0..127
    const int ldK   = (tid & 1) * 16;      // 0 or 16 (bf16 elems)
    const unsigned thLds = (unsigned)((ldRow * LDR + ldK) * 2);          // bytes
    const unsigned gAbase = (unsigned)(((bi * 128 + ldRow) * Dcols + ldK) * 2);
    const unsigned gBbase = (unsigned)(((bj * 128 + ldRow) * Dcols + ldK) * 2);

#if USE_ASYNC_LDS
    const unsigned smemBase = (unsigned)(uintptr_t)(&smem[0][0][0]);
    const unsigned PB = (unsigned)(PANEL * 2);   // panel bytes (10240)
    unsigned ldsOff[2][4];
#pragma unroll
    for (int b = 0; b < 2; ++b)
#pragma unroll
        for (int p = 0; p < 4; ++p)
            ldsOff[b][p] = smemBase + (unsigned)(b * 4 + p) * PB + thLds;

    // prologue: prefetch chunk 0 into stage 0
    async_stage(xhi, xlo, gAbase, gBbase,
                ldsOff[0][0], ldsOff[0][1], ldsOff[0][2], ldsOff[0][3]);
#endif

    for (int i = 0; i < NK; ++i) {
        const int cur = i & 1;
#if USE_ASYNC_LDS
        // drain own async copies for the stage we are about to read,
        // then rendezvous so every wave's stage data is visible.
        asm volatile("s_wait_asynccnt 0x0" ::: "memory");
        __syncthreads();
        if (i + 1 < NK) {   // prefetch next chunk into the other stage
            const unsigned go = (unsigned)((i + 1) * KC * 2);
            const int nxt = cur ^ 1;
            async_stage(xhi, xlo, gAbase + go, gBbase + go,
                        ldsOff[nxt][0], ldsOff[nxt][1],
                        ldsOff[nxt][2], ldsOff[nxt][3]);
        }
#else
        __syncthreads();
        {
            const int kc = i * KC;
            uint4* dA0 = (uint4*)(&smem[cur][0][0] + ldRow * LDR + ldK);
            uint4* dA1 = (uint4*)(&smem[cur][1][0] + ldRow * LDR + ldK);
            uint4* dB0 = (uint4*)(&smem[cur][2][0] + ldRow * LDR + ldK);
            uint4* dB1 = (uint4*)(&smem[cur][3][0] + ldRow * LDR + ldK);
            const uint4* s;
            s = (const uint4*)(xhi + (gAbase >> 1) + kc); dA0[0] = s[0]; dA0[1] = s[1];
            s = (const uint4*)(xlo + (gAbase >> 1) + kc); dA1[0] = s[0]; dA1[1] = s[1];
            s = (const uint4*)(xhi + (gBbase >> 1) + kc); dB0[0] = s[0]; dB0[1] = s[1];
            s = (const uint4*)(xlo + (gBbase >> 1) + kc); dB1[0] = s[0]; dB1[1] = s[1];
        }
        __syncthreads();
#endif
        const __bf16* sAhi = &smem[cur][0][0];
        const __bf16* sAlo = &smem[cur][1][0];
        const __bf16* sBhi = &smem[cur][2][0];
        const __bf16* sBlo = &smem[cur][3][0];

        // A fragments (16x32 bf16): elems 0-7 <- K=lh*8.., elems 8-15 <- K=16+lh*8..
        v16bf aHi[2], aLo[2];
#pragma unroll
        for (int mt = 0; mt < 2; ++mt) {
            const __bf16* rHi = sAhi + (wRow + mt * 16 + n16) * LDR;
            const __bf16* rLo = sAlo + (wRow + mt * 16 + n16) * LDR;
            union { v16bf v; uint4 q[2]; } ua, ub;
            ua.q[0] = *(const uint4*)(rHi + lh * 8);
            ua.q[1] = *(const uint4*)(rHi + 16 + lh * 8);
            ub.q[0] = *(const uint4*)(rLo + lh * 8);
            ub.q[1] = *(const uint4*)(rLo + 16 + lh * 8);
            aHi[mt] = ua.v;
            aLo[mt] = ub.v;
        }

#pragma unroll
        for (int nt = 0; nt < 4; ++nt) {
            // B fragments (32x16 bf16): lane holds column n16, elems 0-7 <- K=lh*8.., elems 8-15 <- K=16+lh*8..
            const __bf16* rHi = sBhi + (wCol + nt * 16 + n16) * LDR;
            const __bf16* rLo = sBlo + (wCol + nt * 16 + n16) * LDR;
            union { v16bf v; uint4 q[2]; } ub, uc;
            ub.q[0] = *(const uint4*)(rHi + lh * 8);
            ub.q[1] = *(const uint4*)(rHi + 16 + lh * 8);
            uc.q[0] = *(const uint4*)(rLo + lh * 8);
            uc.q[1] = *(const uint4*)(rLo + 16 + lh * 8);
            const v16bf bHi = ub.v, bLo = uc.v;
#pragma unroll
            for (int mt = 0; mt < 2; ++mt) {
                acc[mt][nt] = __builtin_amdgcn_wmma_f32_16x16x32_bf16(
                    false, aHi[mt], false, bHi, (short)0, acc[mt][nt], false, false);
                acc[mt][nt] = __builtin_amdgcn_wmma_f32_16x16x32_bf16(
                    false, aHi[mt], false, bLo, (short)0, acc[mt][nt], false, false);
                acc[mt][nt] = __builtin_amdgcn_wmma_f32_16x16x32_bf16(
                    false, aLo[mt], false, bHi, (short)0, acc[mt][nt], false, false);
            }
        }
#if !USE_ASYNC_LDS
#else
        __syncthreads();   // all waves done reading `cur` before it is refilled
#endif
    }

    // --- fused masked epilogue: C layout is m = e + 8*lh, n = lane&15 ---
    float p1 = 0.0f, p2 = 0.0f, p3 = 0.0f;
#pragma unroll
    for (int nt = 0; nt < 4; ++nt) {
        const int j  = (int)bj * 128 + wCol + nt * 16 + n16;
        const int tj = tgt[j];
#pragma unroll
        for (int mt = 0; mt < 2; ++mt) {
            const int ibase = (int)bi * 128 + wRow + mt * 16 + 8 * lh;
            union { v8f v; float f[8]; } u;
            u.v = acc[mt][nt];
#pragma unroll
            for (int e = 0; e < 8; ++e) {
                const int i = ibase + e;
                const float g = u.f[e];
                if (i > j) {
                    p3 += g * g;
                    if (tgt[i] == tj) { p1 += g; p2 += 1.0f; }
                }
            }
        }
    }

    // wave32 shuffle reduction, then LDS atomic combine per block
#pragma unroll
    for (int off = 16; off > 0; off >>= 1) {
        p1 += __shfl_down(p1, off, 32);
        p2 += __shfl_down(p2, off, 32);
        p3 += __shfl_down(p3, off, 32);
    }
    if (lane == 0) {
        atomicAdd(&sAcc[0], p1);   // ds_add_f32 (LDS scope only)
        atomicAdd(&sAcc[1], p2);
        atomicAdd(&sAcc[2], p3);
    }
    __syncthreads();
    if (tid == 0) {
        partials[blockIdx.x * 3 + 0] = sAcc[0];
        partials[blockIdx.x * 3 + 1] = sAcc[1];
        partials[blockIdx.x * 3 + 2] = sAcc[2];
    }
}

// ---------------------------------------------------------------------------
// Kernel 3: reduce 528 partial triples, emit the scalar loss
// ---------------------------------------------------------------------------
__global__ __launch_bounds__(256)
void finalize_loss(const float* __restrict__ partials, int n,
                   float* __restrict__ out) {
    __shared__ float s1[8], s2[8], s3[8];
    float a = 0.0f, b = 0.0f, c = 0.0f;
    for (int i = threadIdx.x; i < n; i += 256) {
        a += partials[3 * i + 0];
        b += partials[3 * i + 1];
        c += partials[3 * i + 2];
    }
#pragma unroll
    for (int off = 16; off > 0; off >>= 1) {
        a += __shfl_down(a, off, 32);
        b += __shfl_down(b, off, 32);
        c += __shfl_down(c, off, 32);
    }
    const int wave = threadIdx.x >> 5;
    if ((threadIdx.x & 31) == 0) { s1[wave] = a; s2[wave] = b; s3[wave] = c; }
    __syncthreads();
    if (threadIdx.x == 0) {
        float A = 0.0f, B = 0.0f, C = 0.0f;
#pragma unroll
        for (int w = 0; w < 8; ++w) { A += s1[w]; B += s2[w]; C += s3[w]; }
        out[0] = -(A / (B * sqrtf(C)));
    }
}

// ---------------------------------------------------------------------------
extern "C" void kernel_launch(void* const* d_in, const int* in_sizes, int n_in,
                              void* d_out, int out_size, void* d_ws, size_t ws_size,
                              hipStream_t stream) {
    const float* x   = (const float*)d_in[0];   // [4096, 1024] f32
    const int*   tgt = (const int*)d_in[1];     // [4096] int32 (JAX default x64-off)
    float*       out = (float*)d_out;           // scalar

    const size_t nElem = (size_t)Nrows * Dcols;
    __bf16* xhi      = (__bf16*)d_ws;
    __bf16* xlo      = xhi + nElem;
    float*  partials = (float*)(xlo + nElem);   // 528*3 floats

    // 1) split f32 -> bf16 hi/lo (nElem/4 float4 work-items)
    convert_split<<<dim3((unsigned)(nElem / 4 / 256)), dim3(256), 0, stream>>>(
        x, xhi, xlo);

    // 2) 528 lower-triangular 128x128 tiles, fused masked reduction
    gram_tril_wmma<<<dim3(NTILE), dim3(256), 0, stream>>>(xhi, xlo, tgt, partials);

    // 3) final scalar
    finalize_loss<<<dim3(1), dim3(256), 0, stream>>>(partials, NTILE, out);
}
